// SpikingConvStem_73254962200916
// MI455X (gfx1250) — matrix operand order repaired and checked by reading, not compiled, not run on hardware
//
#include <hip/hip_runtime.h>

typedef __attribute__((ext_vector_type(16))) _Float16 v16h;
typedef __attribute__((ext_vector_type(8)))  _Float16 v8h;
typedef __attribute__((ext_vector_type(8)))  float    v8f;

#define T_   4
#define B_   64
#define CIN  3
#define H0   64
#define W0   64
#define HID  128
#define H1   32
#define W1   32
#define EMB  256
#define K2   (HID * 9)   /* 1152 */
#define LN_EPS 1e-5f

// workspace layout
#define WS_SPIKE_OFF  (1u << 20)                 /* spikes at +1 MB */
#define WS_ZPAD_OFF   (WS_SPIKE_OFF - 512u)      /* 256-half zero pad */

// ---------------------------------------------------------------------------
// Kernel 0: conv2 weights [oc][ic][kh][kw] f32 -> [oc][(kh*3+kw)*128+ic] f16
// (B^T row-major so each WMMA B-fragment lane reads contiguous halfs).
// Also zeroes the 256-half padding region used for conv-border A loads.
// ---------------------------------------------------------------------------
__global__ __launch_bounds__(256) void prep_w2(const float* __restrict__ w2,
                                               _Float16* __restrict__ o,
                                               _Float16* __restrict__ zpad) {
    int id = blockIdx.x * 256 + threadIdx.x;
    if (id < 256) zpad[id] = (_Float16)0.f;      // guaranteed-zero border pad
    if (id >= EMB * K2) return;
    int oc  = id / K2;
    int r   = id % K2;
    int seg = r >> 7;       // kh*3+kw
    int ic  = r & 127;
    int kh  = seg / 3, kw = seg % 3;
    o[id] = (_Float16)w2[((oc * HID + ic) * 3 + kh) * 3 + kw];
}

// ---------------------------------------------------------------------------
// Kernel 1: conv1 (3->128, 3x3, s2, p1) + LIF over T. Block = one (b,h,w),
// 128 threads = output channels. Spikes stored channels-last f16.
// ---------------------------------------------------------------------------
__global__ __launch_bounds__(128) void conv1_lif(const float* __restrict__ x,
                                                 const float* __restrict__ w1,
                                                 const float* __restrict__ b1,
                                                 _Float16* __restrict__ sp) {
    __shared__ float xs[T_][28];
    const int bi = blockIdx.x;
    const int b  = bi >> 10;
    const int hw = bi & 1023;
    const int h  = hw >> 5, w = hw & 31;
    const int tid = threadIdx.x;

    if (tid < T_ * 27) {
        int t = tid / 27, r = tid % 27;
        int ci = r / 9, kh = (r % 9) / 3, kw = r % 3;
        int ih = 2 * h - 1 + kh, iw = 2 * w - 1 + kw;   // max 63, only -1 OOB
        float v = 0.f;
        if (ih >= 0 && iw >= 0)
            v = x[(((size_t)(t * B_ + b) * CIN + ci) * H0 + ih) * W0 + iw];
        xs[t][r] = v;
    }
    __syncthreads();

    const int c = tid;
    float wreg[27];
#pragma unroll
    for (int i = 0; i < 27; ++i) wreg[i] = w1[c * 27 + i];
    const float bias = b1[c];

    float v = 0.f;
#pragma unroll
    for (int t = 0; t < T_; ++t) {
        float acc = bias;
#pragma unroll
        for (int i = 0; i < 27; ++i) acc += xs[t][i] * wreg[i];
        v = 0.5f * (v + acc);                 // v += (x - v)/tau, tau=2
        float s = (v >= 1.f) ? 1.f : 0.f;     // fire
        v = (s > 0.f) ? 0.f : v;              // hard reset
        sp[((size_t)(t * B_ + b) * 1024 + hw) * HID + c] = (_Float16)s;
    }
}

// ---------------------------------------------------------------------------
// Kernel 2: conv2 as WMMA GEMM. M=65536 (t,b,oh,ow), N=256 (oc), K=1152.
// Block: 8 waves, 64x256 tile. Wave = 2 M-tiles x 4 N-tiles (acc 64 VGPRs).
// No LDS / no barriers. A fragments load straight from the L2-resident spike
// image in contiguous k order (k = 16h+s); conv-border lanes read a zero pad
// region instead of predicating, so EXEC stays all-ones and no zero-fill
// VALU ops are needed. K loop nested (kh,kw,ic-chunk): segment math invariant.
// ---------------------------------------------------------------------------
__global__ __launch_bounds__(256) void conv2_wmma(const _Float16* __restrict__ sp,
                                                  const _Float16* __restrict__ zpad,
                                                  const _Float16* __restrict__ w2,
                                                  const float* __restrict__ b2,
                                                  float* __restrict__ out) {
    const int tid  = threadIdx.x;
    const int lane = tid & 31;
    const int wave = tid >> 5;
    const int wmh  = wave & 1;         // which pair of M-tiles (32 rows)
    const int wnq  = wave >> 1;        // N quarter (64 cols)
    const int img  = blockIdx.x >> 2;  // t*64+b
    const int n0   = (blockIdx.x & 3) * 64;

    // A fragment rows for this lane, one per M-tile mt=0,1
    // lane<16 -> K{0-15}, lane>=16 -> K{16-31}
    const int akoff = (lane >> 4) << 4;          // 0 or 16 halfs
    int oh[2], ow[2];
#pragma unroll
    for (int mt = 0; mt < 2; ++mt) {
        int n = n0 + wmh * 32 + mt * 16 + (lane & 15);
        oh[mt] = n >> 4;
        ow[mt] = n & 15;
    }
    const _Float16* img_base = sp + (size_t)img * 1024 * HID;

    // B fragment: lane holds 16 contiguous K of column N = lane&15
    const int ocb = wnq * 64 + (lane & 15);
    const int bko = (lane >> 4) << 4;            // K offset 0 or 16

    v8f acc[2][4];
#pragma unroll
    for (int mt = 0; mt < 2; ++mt)
#pragma unroll
        for (int nt = 0; nt < 4; ++nt)
#pragma unroll
            for (int j = 0; j < 8; ++j) acc[mt][nt][j] = 0.f;

#pragma unroll 1
    for (int kh = 0; kh < 3; ++kh) {
#pragma unroll 1
        for (int kw = 0; kw < 3; ++kw) {
            const int kbase = (kh * 3 + kw) * 128;
            // per-M-tile source line; border lanes point at the zero pad
            const _Float16* ap[2];
#pragma unroll
            for (int mt = 0; mt < 2; ++mt) {
                int ih = 2 * oh[mt] - 1 + kh;    // in [-1,31]
                int iw = 2 * ow[mt] - 1 + kw;    // in [-1,31]
                bool valid = (ih >= 0) && (iw >= 0);
                const _Float16* base =
                    valid ? (img_base + (ih * 32 + iw) * HID) : zpad;
                ap[mt] = base + akoff;
            }
#pragma unroll
            for (int kc = 0; kc < 4; ++kc) {     // 32-K chunks of the segment
                const int k0 = kbase + kc * 32;
                // A fragments: unconditional dual b128 per lane
                union { v16h v; v8h h[2]; } a[2];
#pragma unroll
                for (int mt = 0; mt < 2; ++mt) {
                    a[mt].h[0] = *(const v8h*)(ap[mt] + kc * 32);
                    a[mt].h[1] = *(const v8h*)(ap[mt] + kc * 32 + 8);
                }
                // B fragments into distinct registers -> one load clause,
                // partial waits overlap L2 fetch with the WMMA chain
                v16h bfrag[4];
#pragma unroll
                for (int nt = 0; nt < 4; ++nt)
                    bfrag[nt] = *(const v16h*)(w2 + (size_t)(ocb + nt * 16) * K2
                                               + k0 + bko);
#pragma unroll
                for (int mt = 0; mt < 2; ++mt)
#pragma unroll
                    for (int nt = 0; nt < 4; ++nt)
                        acc[mt][nt] = __builtin_amdgcn_wmma_f32_16x16x32_f16(
                            false, a[mt].v, false, bfrag[nt], (short)0,
                            acc[mt][nt], false, false);
            }
        }
    }

    // epilogue: fused bias, store directly in final (t,b,n,d) layout
    const int mbase = (lane >> 4) ? 8 : 0;
#pragma unroll
    for (int mt = 0; mt < 2; ++mt)
#pragma unroll
        for (int nt = 0; nt < 4; ++nt) {
            int col = wnq * 64 + nt * 16 + (lane & 15);
            float bias = b2[col];
#pragma unroll
            for (int j = 0; j < 8; ++j) {
                int row = n0 + wmh * 32 + mt * 16 + mbase + j;
                out[((size_t)img * 256 + row) * EMB + col] = acc[mt][nt][j] + bias;
            }
        }
}

// ---------------------------------------------------------------------------
// Kernel 3: LIF over T + LayerNorm over D=256, in place on d_out.
// Block = one (b,n); thread = d. Spikes are Bernoulli -> var = m - m^2.
// ---------------------------------------------------------------------------
__global__ __launch_bounds__(256) void lif_ln(float* __restrict__ io,
                                              const float* __restrict__ g,
                                              const float* __restrict__ bb) {
    __shared__ float red[256];
    const int bn = blockIdx.x;                   // b*256+n
    const int d  = threadIdx.x;
    const size_t TS   = (size_t)B_ * 256 * EMB;  // stride between t slices
    const size_t base = (size_t)bn * EMB + d;

    float s[T_];
    float v = 0.f;
#pragma unroll
    for (int t = 0; t < T_; ++t) {
        float xx = io[t * TS + base];
        v = 0.5f * (v + xx);
        float sp = (v >= 1.f) ? 1.f : 0.f;
        v = (sp > 0.f) ? 0.f : v;
        s[t] = sp;
    }
    const float gd = g[d], bd = bb[d];
#pragma unroll
    for (int t = 0; t < T_; ++t) {
        red[d] = s[t];
        __syncthreads();
        for (int off = 128; off > 0; off >>= 1) {
            if (d < off) red[d] += red[d + off];
            __syncthreads();
        }
        float m = red[0] * (1.f / 256.f);
        __syncthreads();
        float var = m - m * m;                    // E[s^2]=E[s] for s in {0,1}
        io[t * TS + base] = (s[t] - m) * rsqrtf(var + LN_EPS) * gd + bd;
    }
}

// ---------------------------------------------------------------------------
extern "C" void kernel_launch(void* const* d_in, const int* in_sizes, int n_in,
                              void* d_out, int out_size, void* d_ws, size_t ws_size,
                              hipStream_t stream) {
    (void)in_sizes; (void)n_in; (void)out_size; (void)ws_size;
    const float* x  = (const float*)d_in[0];
    const float* w1 = (const float*)d_in[1];
    const float* b1 = (const float*)d_in[2];
    const float* w2 = (const float*)d_in[3];
    const float* b2 = (const float*)d_in[4];
    const float* lg = (const float*)d_in[5];
    const float* lb = (const float*)d_in[6];
    float* out = (float*)d_out;

    _Float16* w2h  = (_Float16*)d_ws;                            // 576 KB
    _Float16* zpad = (_Float16*)((char*)d_ws + WS_ZPAD_OFF);     // 512 B zeros
    _Float16* sp   = (_Float16*)((char*)d_ws + WS_SPIKE_OFF);    // 64 MB spikes

    prep_w2  <<<(EMB * K2 + 255) / 256, 256, 0, stream>>>(w2, w2h, zpad);
    conv1_lif<<<B_ * H1 * W1,           128, 0, stream>>>(x, w1, b1, sp);
    conv2_wmma<<<T_ * B_ * 4,           256, 0, stream>>>(sp, zpad, w2h, b2, out);
    lif_ln   <<<B_ * 256,               256, 0, stream>>>(out, lg, lb);
}
